// FlashDeepseekLayer_86500641341792
// MI455X (gfx1250) — matrix operand order repaired and checked by reading, not compiled, not run on hardware
//
#include <hip/hip_runtime.h>
#include <hip/hip_bf16.h>

// ---------------------------------------------------------------------------
// MoE layer for MI455X (gfx1250): bf16 WMMA (v_wmma_f32_16x16x32_bf16), f32
// accumulate. f32 operands are converted to bf16 while staging into LDS.
// B tiles are stored transposed [N][K] in LDS (register 4x4 micro-transpose)
// so every WMMA fragment read is a contiguous 16B ds_load_b128.
// ---------------------------------------------------------------------------

typedef __attribute__((ext_vector_type(16))) __bf16 v16bf;
typedef __attribute__((ext_vector_type(8)))  float  v8f;

#define T_TOK 2048
#define DDIM  2048
#define EEXP  64
#define KTOP  6
#define FDIM  1408
#define FSDIM 2816
#define CAP   512

// Block: 256 threads = 8 waves in a 2(M) x 4(N) grid.
// Block tile: M=32 x N=128, K-step 32. Wave tile: 16 x 32 (2 B-frags).
#define TM 32
#define TN 128
#define TK 32
#define LDK 40  // padded K-stride (bf16) for LDS tiles (80B: 16B-aligned rows)

__device__ __forceinline__ float silu_f(float v) {
  return v * __builtin_amdgcn_rcpf(1.0f + __expf(-v));
}

__device__ __forceinline__ v8f wmma_bf16(v16bf a, v16bf b, v8f c) {
  return __builtin_amdgcn_wmma_f32_16x16x32_bf16(
      /*neg_a=*/false, a, /*neg_b=*/false, b,
      /*c_mod=*/(short)0, c, /*reuse_a=*/false, /*reuse_b=*/false);
}

// A fragment (16x32 bf16, ISA 7.12.2): lane<16 row M=lr holds K={0..7} in
// elems 0..7, K={16..23} in elems 8..15; lanes>=16 shift K by +8.
// Ash is [M][LDK] (K-contiguous) -> two contiguous 16B chunks per lane.
__device__ __forceinline__ v16bf load_a_frag(__bf16 (*Ash)[LDK], int row, int hi) {
  v16bf a;
  const __bf16* p0 = &Ash[row][hi * 8];
  const __bf16* p1 = &Ash[row][16 + hi * 8];
#pragma unroll
  for (int j = 0; j < 8; ++j) a[j] = p0[j];
#pragma unroll
  for (int j = 0; j < 8; ++j) a[8 + j] = p1[j];
  return a;
}

// B fragment (32x16 bf16): lane<16 col N=lr holds K={0..7} in elems 0..7,
// K={16..23} in elems 8..15; lanes>=16 shift K by +8.
// Bsh is transposed [N][LDK] -> two contiguous 16B chunks per lane.
__device__ __forceinline__ v16bf load_b_frag(__bf16 (*Bsh)[LDK], int col, int hi) {
  v16bf b;
  const __bf16* p0 = &Bsh[col][hi * 8];
  const __bf16* p1 = &Bsh[col][16 + hi * 8];
#pragma unroll
  for (int j = 0; j < 8; ++j) b[j] = p0[j];
#pragma unroll
  for (int j = 0; j < 8; ++j) b[8 + j] = p1[j];
  return b;
}

// --- B tile staging: 32(K) x 128(N) f32 -> bf16 LDS transposed [N][K]. ----
// Each thread owns one 4x4 micro-block: kb = t&7 (K group), nb = t>>3 (N group).
struct BReg { float4 r[4]; };

__device__ __forceinline__ void ldg_b(const float* __restrict__ src, size_t ld,
                                      int kb, int nb, BReg& b) {
#pragma unroll
  for (int i = 0; i < 4; ++i)
    b.r[i] = *(const float4*)(src + (size_t)(kb * 4 + i) * ld + nb * 4);
}

__device__ __forceinline__ void st_b(__bf16 (*Bsh)[LDK], int kb, int nb,
                                     const BReg& b) {
#pragma unroll
  for (int c = 0; c < 4; ++c) {
    union { __bf16 h[4]; uint2 u; } col;
    col.h[0] = (__bf16)((const float*)&b.r[0])[c];
    col.h[1] = (__bf16)((const float*)&b.r[1])[c];
    col.h[2] = (__bf16)((const float*)&b.r[2])[c];
    col.h[3] = (__bf16)((const float*)&b.r[3])[c];
    *(uint2*)&Bsh[nb * 4 + c][kb * 4] = col.u;  // conflict-free across a wave
  }
}

// A tile staging (f32 source): thread ai = t>>3 (row), ac = t&7 (float4 col).
__device__ __forceinline__ void st_a_f32(__bf16 (*Ash)[LDK], int ai, int ac,
                                         float4 v) {
  union { __bf16 h[4]; uint2 u; } q;
  q.h[0] = (__bf16)v.x; q.h[1] = (__bf16)v.y;
  q.h[2] = (__bf16)v.z; q.h[3] = (__bf16)v.w;
  *(uint2*)&Ash[ai][ac * 4] = q.u;
}

// ---------------------------------------------------------------------------
// 1) Gate: logits = x @ gate_w^T, softmax, top-6, renormalize.
// ---------------------------------------------------------------------------
__global__ __launch_bounds__(256)
void gate_topk_kernel(const float* __restrict__ x, const float* __restrict__ gw,
                      int* __restrict__ topi, float* __restrict__ topw) {
  __shared__ float part[256];
  __shared__ float logit[EEXP];
  const int t = blockIdx.x;
  const int tid = threadIdx.x;
  const int e = tid & 63;
  const int chunk = tid >> 6;
  const float* xr = x + (size_t)t * DDIM + chunk * 512;
  const float* gr = gw + (size_t)e * DDIM + chunk * 512;
  float s = 0.f;
#pragma unroll 4
  for (int d = 0; d < 512; d += 4) {
    float4 a = *(const float4*)(xr + d);
    float4 b = *(const float4*)(gr + d);
    s += a.x * b.x + a.y * b.y + a.z * b.z + a.w * b.w;
  }
  part[tid] = s;
  __syncthreads();
  if (tid < EEXP)
    logit[tid] = part[tid] + part[tid + 64] + part[tid + 128] + part[tid + 192];
  __syncthreads();
  if (tid == 0) {
    float mx = logit[0];
    for (int i = 1; i < EEXP; ++i) mx = fmaxf(mx, logit[i]);
    for (int i = 0; i < EEXP; ++i) logit[i] = __expf(logit[i] - mx);
    // softmax denominator cancels under top-k renormalization
    int bi[KTOP]; float bv[KTOP]; float wsum = 0.f;
    for (int k = 0; k < KTOP; ++k) {
      float best = -1.f; int bidx = 0;
      for (int i = 0; i < EEXP; ++i)
        if (logit[i] > best) { best = logit[i]; bidx = i; }
      logit[bidx] = -2.f;
      bi[k] = bidx; bv[k] = best; wsum += best;
    }
    float inv = 1.f / wsum;
    for (int k = 0; k < KTOP; ++k) {
      topi[t * KTOP + k] = bi[k];
      topw[t * KTOP + k] = bv[k] * inv;
    }
  }
}

__global__ void init_cnt_kernel(int* __restrict__ cnt) {
  if (threadIdx.x < EEXP) cnt[threadIdx.x] = 0;
}

// ---------------------------------------------------------------------------
// 2) Dispatch: unique slot per (token, expert) assignment via atomics.
// ---------------------------------------------------------------------------
__global__ __launch_bounds__(256)
void dispatch_kernel(const int* __restrict__ topi, const float* __restrict__ topw,
                     int* __restrict__ cnt, int* __restrict__ rows,
                     float* __restrict__ wts) {
  int a = blockIdx.x * 256 + threadIdx.x;
  if (a >= T_TOK * KTOP) return;
  int e = topi[a];
  int slot = atomicAdd(&cnt[e], 1);
  if (slot < CAP) {
    rows[e * CAP + slot] = a / KTOP;
    wts[e * CAP + slot]  = topw[a];
  }
}

// ---------------------------------------------------------------------------
// 3) Expert gate/up GEMMs + SwiGLU (A rows gathered via per-expert lists).
// ---------------------------------------------------------------------------
__global__ __launch_bounds__(256)
void expert_gateup_kernel(const float* __restrict__ x,
                          const float* __restrict__ wg,
                          const float* __restrict__ wu,
                          const int* __restrict__ cnt,
                          const int* __restrict__ rows,
                          __bf16* __restrict__ hbuf) {
  const int e  = blockIdx.z;
  const int mt = blockIdx.y;
  const int f0 = blockIdx.x * TN;
  const int n_e = min(cnt[e], CAP);
  const int m_base = mt * TM;
  if (m_base >= n_e) return;

  __shared__ __align__(16) __bf16 Ash[TM][LDK];
  __shared__ __align__(16) __bf16 Bgs[TN][LDK];
  __shared__ __align__(16) __bf16 Bus[TN][LDK];

  const int tid = threadIdx.x;
  const int lane = tid & 31, w = tid >> 5;
  const int wm = w & 1, wn = w >> 1;
  const int lr = lane & 15, hi = lane >> 4;
  const int ai = tid >> 3, ac = tid & 7;  // A staging
  const int kb = tid & 7, nb = tid >> 3;  // B staging (4x4 micro-transpose)

  const float* aptr = nullptr;
  if (m_base + ai < n_e) {
    int tok = rows[e * CAP + m_base + ai];
    aptr = x + (size_t)tok * DDIM + ac * 4;
  }
  const size_t wbase = (size_t)e * DDIM * FDIM + f0;

  // prologue: prefetch k0 = 0 into registers
  float4 aReg = make_float4(0.f, 0.f, 0.f, 0.f);
  if (aptr) aReg = *(const float4*)(aptr);
  BReg bgReg, buReg;
  ldg_b(wg + wbase, FDIM, kb, nb, bgReg);
  ldg_b(wu + wbase, FDIM, kb, nb, buReg);

  v8f accg[2] = {{}, {}};
  v8f accu[2] = {{}, {}};

  for (int k0 = 0; k0 < DDIM; k0 += TK) {
    st_a_f32(Ash, ai, ac, aReg);
    st_b(Bgs, kb, nb, bgReg);
    st_b(Bus, kb, nb, buReg);
    __syncthreads();

    if (k0 + TK < DDIM) {  // prefetch next K-tile during compute
      aReg = make_float4(0.f, 0.f, 0.f, 0.f);
      if (aptr) aReg = *(const float4*)(aptr + k0 + TK);
      ldg_b(wg + wbase + (size_t)(k0 + TK) * FDIM, FDIM, kb, nb, bgReg);
      ldg_b(wu + wbase + (size_t)(k0 + TK) * FDIM, FDIM, kb, nb, buReg);
    }

    v16bf a = load_a_frag(Ash, wm * 16 + lr, hi);
#pragma unroll
    for (int sub = 0; sub < 2; ++sub) {
      int coln = wn * 32 + sub * 16 + lr;
      v16bf bg = load_b_frag(Bgs, coln, hi);
      v16bf bu = load_b_frag(Bus, coln, hi);
      accg[sub] = wmma_bf16(a, bg, accg[sub]);
      accu[sub] = wmma_bf16(a, bu, accu[sub]);
    }
    __syncthreads();
  }

#pragma unroll
  for (int sub = 0; sub < 2; ++sub) {
    const int col = f0 + wn * 32 + sub * 16 + lr;
#pragma unroll
    for (int r = 0; r < 8; ++r) {
      int m = m_base + wm * 16 + r + hi * 8;
      if (m < n_e) {
        float h = silu_f(accg[sub][r]) * accu[sub][r];
        hbuf[((size_t)e * CAP + m) * FDIM + col] = (__bf16)h;
      }
    }
  }
}

// ---------------------------------------------------------------------------
// 4) Expert down GEMM + weighted combine: y[tok] += wt * (h @ w_down).
// ---------------------------------------------------------------------------
__global__ __launch_bounds__(256)
void expert_down_kernel(const __bf16* __restrict__ hbuf,
                        const float* __restrict__ wd,
                        const int* __restrict__ cnt,
                        const int* __restrict__ rows,
                        const float* __restrict__ wts,
                        float* __restrict__ y) {
  const int e  = blockIdx.z;
  const int mt = blockIdx.y;
  const int d0 = blockIdx.x * TN;
  const int n_e = min(cnt[e], CAP);
  const int m_base = mt * TM;
  if (m_base >= n_e) return;

  __shared__ __align__(16) __bf16 Ash[TM][LDK];
  __shared__ __align__(16) __bf16 Bds[TN][LDK];

  const int tid = threadIdx.x;
  const int lane = tid & 31, w = tid >> 5;
  const int wm = w & 1, wn = w >> 1;
  const int lr = lane & 15, hi = lane >> 4;
  const int ai = tid >> 3, ac = tid & 7;
  const int kb = tid & 7, nb = tid >> 3;

  // rows >= n_e read garbage; results masked in the epilogue
  const __bf16* asrc = hbuf + ((size_t)e * CAP + m_base + ai) * FDIM + ac * 4;
  const size_t wbase = (size_t)e * FDIM * DDIM + d0;

  uint2 aReg = *(const uint2*)(asrc);
  BReg bReg;
  ldg_b(wd + wbase, DDIM, kb, nb, bReg);

  v8f acc[2] = {{}, {}};

  for (int k0 = 0; k0 < FDIM; k0 += TK) {
    *(uint2*)&Ash[ai][ac * 4] = aReg;
    st_b(Bds, kb, nb, bReg);
    __syncthreads();

    if (k0 + TK < FDIM) {
      aReg = *(const uint2*)(asrc + k0 + TK);
      ldg_b(wd + wbase + (size_t)(k0 + TK) * DDIM, DDIM, kb, nb, bReg);
    }

    v16bf a = load_a_frag(Ash, wm * 16 + lr, hi);
#pragma unroll
    for (int sub = 0; sub < 2; ++sub) {
      v16bf b = load_b_frag(Bds, wn * 32 + sub * 16 + lr, hi);
      acc[sub] = wmma_bf16(a, b, acc[sub]);
    }
    __syncthreads();
  }

#pragma unroll
  for (int sub = 0; sub < 2; ++sub) {
    const int col = d0 + wn * 32 + sub * 16 + lr;
#pragma unroll
    for (int r = 0; r < 8; ++r) {
      int m = m_base + wm * 16 + r + hi * 8;
      if (m < n_e) {
        int tok  = rows[e * CAP + m];
        float wt = wts[e * CAP + m];
        atomicAdd(&y[(size_t)tok * DDIM + col], wt * acc[sub][r]);
      }
    }
  }
}

// ---------------------------------------------------------------------------
// 5) Shared expert gate/up: hs = silu(x@wsg) * (x@wsu), bf16.
// ---------------------------------------------------------------------------
__global__ __launch_bounds__(256)
void shared_gateup_kernel(const float* __restrict__ x,
                          const float* __restrict__ wsg,
                          const float* __restrict__ wsu,
                          __bf16* __restrict__ hs) {
  const int t0 = blockIdx.y * TM;
  const int f0 = blockIdx.x * TN;

  __shared__ __align__(16) __bf16 Ash[TM][LDK];
  __shared__ __align__(16) __bf16 Bgs[TN][LDK];
  __shared__ __align__(16) __bf16 Bus[TN][LDK];

  const int tid = threadIdx.x;
  const int lane = tid & 31, w = tid >> 5;
  const int wm = w & 1, wn = w >> 1;
  const int lr = lane & 15, hi = lane >> 4;
  const int ai = tid >> 3, ac = tid & 7;
  const int kb = tid & 7, nb = tid >> 3;

  const float* aptr = x + (size_t)(t0 + ai) * DDIM + ac * 4;

  float4 aReg = *(const float4*)(aptr);
  BReg bgReg, buReg;
  ldg_b(wsg + f0, FSDIM, kb, nb, bgReg);
  ldg_b(wsu + f0, FSDIM, kb, nb, buReg);

  v8f accg[2] = {{}, {}};
  v8f accu[2] = {{}, {}};

  for (int k0 = 0; k0 < DDIM; k0 += TK) {
    st_a_f32(Ash, ai, ac, aReg);
    st_b(Bgs, kb, nb, bgReg);
    st_b(Bus, kb, nb, buReg);
    __syncthreads();

    if (k0 + TK < DDIM) {
      aReg = *(const float4*)(aptr + k0 + TK);
      ldg_b(wsg + (size_t)(k0 + TK) * FSDIM + f0, FSDIM, kb, nb, bgReg);
      ldg_b(wsu + (size_t)(k0 + TK) * FSDIM + f0, FSDIM, kb, nb, buReg);
    }

    v16bf a = load_a_frag(Ash, wm * 16 + lr, hi);
#pragma unroll
    for (int sub = 0; sub < 2; ++sub) {
      int coln = wn * 32 + sub * 16 + lr;
      v16bf bg = load_b_frag(Bgs, coln, hi);
      v16bf bu = load_b_frag(Bus, coln, hi);
      accg[sub] = wmma_bf16(a, bg, accg[sub]);
      accu[sub] = wmma_bf16(a, bu, accu[sub]);
    }
    __syncthreads();
  }

#pragma unroll
  for (int sub = 0; sub < 2; ++sub) {
    const int col = f0 + wn * 32 + sub * 16 + lr;
#pragma unroll
    for (int r = 0; r < 8; ++r) {
      int m = t0 + wm * 16 + r + hi * 8;
      float h = silu_f(accg[sub][r]) * accu[sub][r];
      hs[(size_t)m * FSDIM + col] = (__bf16)h;
    }
  }
}

// ---------------------------------------------------------------------------
// 6) Shared expert down: y = hs @ ws_down (plain stores; runs before combine).
// ---------------------------------------------------------------------------
__global__ __launch_bounds__(256)
void shared_down_kernel(const __bf16* __restrict__ hs,
                        const float* __restrict__ wsd,
                        float* __restrict__ y) {
  const int t0 = blockIdx.y * TM;
  const int d0 = blockIdx.x * TN;

  __shared__ __align__(16) __bf16 Ash[TM][LDK];
  __shared__ __align__(16) __bf16 Bds[TN][LDK];

  const int tid = threadIdx.x;
  const int lane = tid & 31, w = tid >> 5;
  const int wm = w & 1, wn = w >> 1;
  const int lr = lane & 15, hi = lane >> 4;
  const int ai = tid >> 3, ac = tid & 7;
  const int kb = tid & 7, nb = tid >> 3;

  const __bf16* asrc = hs + (size_t)(t0 + ai) * FSDIM + ac * 4;

  uint2 aReg = *(const uint2*)(asrc);
  BReg bReg;
  ldg_b(wsd + d0, DDIM, kb, nb, bReg);

  v8f acc[2] = {{}, {}};

  for (int k0 = 0; k0 < FSDIM; k0 += TK) {
    *(uint2*)&Ash[ai][ac * 4] = aReg;
    st_b(Bds, kb, nb, bReg);
    __syncthreads();

    if (k0 + TK < FSDIM) {
      aReg = *(const uint2*)(asrc + k0 + TK);
      ldg_b(wsd + (size_t)(k0 + TK) * DDIM + d0, DDIM, kb, nb, bReg);
    }

    v16bf a = load_a_frag(Ash, wm * 16 + lr, hi);
#pragma unroll
    for (int sub = 0; sub < 2; ++sub) {
      v16bf b = load_b_frag(Bds, wn * 32 + sub * 16 + lr, hi);
      acc[sub] = wmma_bf16(a, b, acc[sub]);
    }
    __syncthreads();
  }

#pragma unroll
  for (int sub = 0; sub < 2; ++sub) {
    const int col = d0 + wn * 32 + sub * 16 + lr;
#pragma unroll
    for (int r = 0; r < 8; ++r) {
      int m = t0 + wm * 16 + r + hi * 8;
      y[(size_t)m * DDIM + col] = acc[sub][r];
    }
  }
}

// ---------------------------------------------------------------------------
extern "C" void kernel_launch(void* const* d_in, const int* in_sizes, int n_in,
                              void* d_out, int out_size, void* d_ws, size_t ws_size,
                              hipStream_t stream) {
  const float* x   = (const float*)d_in[0];  // [T, D]
  const float* gw  = (const float*)d_in[1];  // [E, D]
  const float* wg  = (const float*)d_in[2];  // [E, D, F]
  const float* wu  = (const float*)d_in[3];  // [E, D, F]
  const float* wd  = (const float*)d_in[4];  // [E, F, D]
  const float* wsg = (const float*)d_in[5];  // [D, FS]
  const float* wsu = (const float*)d_in[6];  // [D, FS]
  const float* wsd = (const float*)d_in[7];  // [FS, D]
  float* y = (float*)d_out;                  // [T, D]

  char* ws = (char*)d_ws;
  size_t off = 0;
  auto alloc = [&](size_t bytes) {
    void* p = ws + off;
    off += (bytes + 255) & ~(size_t)255;
    return p;
  };
  int*    topi = (int*)   alloc((size_t)T_TOK * KTOP * sizeof(int));
  float*  topw = (float*) alloc((size_t)T_TOK * KTOP * sizeof(float));
  int*    cnt  = (int*)   alloc((size_t)EEXP * sizeof(int));
  int*    rows = (int*)   alloc((size_t)EEXP * CAP * sizeof(int));
  float*  wts  = (float*) alloc((size_t)EEXP * CAP * sizeof(float));
  __bf16* hbuf = (__bf16*)alloc((size_t)EEXP * CAP * FDIM * sizeof(__bf16));
  __bf16* hs   = (__bf16*)alloc((size_t)T_TOK * FSDIM * sizeof(__bf16));
  (void)ws_size; (void)in_sizes; (void)n_in; (void)out_size;

  gate_topk_kernel<<<T_TOK, 256, 0, stream>>>(x, gw, topi, topw);
  init_cnt_kernel<<<1, 64, 0, stream>>>(cnt);
  dispatch_kernel<<<(T_TOK * KTOP + 255) / 256, 256, 0, stream>>>(topi, topw, cnt, rows, wts);

  // shared expert writes y first (plain stores), experts then atomically add
  shared_gateup_kernel<<<dim3(FSDIM / TN, T_TOK / TM), 256, 0, stream>>>(x, wsg, wsu, hs);
  shared_down_kernel<<<dim3(DDIM / TN, T_TOK / TM), 256, 0, stream>>>(hs, wsd, y);

  // expert FFN; blockIdx.z = expert so same-expert blocks are contiguous (L2 reuse)
  expert_gateup_kernel<<<dim3(FDIM / TN, CAP / TM, EEXP), 256, 0, stream>>>(x, wg, wu, cnt, rows, hbuf);
  expert_down_kernel<<<dim3(DDIM / TN, CAP / TM, EEXP), 256, 0, stream>>>(hbuf, wd, cnt, rows, wts, y);
}
